// GraphConv_22213570855128
// MI455X (gfx1250) — matrix operand order repaired and checked by reading, not compiled, not run on hardware
//
#include <hip/hip_runtime.h>
#include <hip/hip_bf16.h>

typedef __attribute__((ext_vector_type(16))) _Float16 v16h;
typedef __attribute__((ext_vector_type(8)))  _Float16 v8h;
typedef __attribute__((ext_vector_type(8)))  float    v8f;

#define NGRAPHS 128

// ---------------- elementwise / setup kernels ----------------

__global__ __launch_bounds__(256) void zero_f32(float* p, long n) {
    long t = (long)blockIdx.x * blockDim.x + threadIdx.x;
    if (t < n) p[t] = 0.0f;
}

__global__ __launch_bounds__(256) void degree_kernel(const int* __restrict__ src,
                                                     const int* __restrict__ dst,
                                                     float* degO, float* degI, int ne) {
    int t = blockIdx.x * blockDim.x + threadIdx.x;
    if (t < ne) {
        atomicAdd(&degO[src[t]], 1.0f);
        atomicAdd(&degI[dst[t]], 1.0f);
    }
}

__global__ __launch_bounds__(256) void rsqrt_clip(float* p, long n) {
    long t = (long)blockIdx.x * blockDim.x + threadIdx.x;
    if (t < n) p[t] = rsqrtf(fmaxf(p[t], 1.0f));
}

// xs[n,f] = x[n,f] * invOut[n]   (D = 69)
__global__ __launch_bounds__(256) void scale_rows69(const float* __restrict__ x,
                                                    const float* __restrict__ invOut,
                                                    float* __restrict__ xs, long total) {
    long t = (long)blockIdx.x * blockDim.x + threadIdx.x;
    if (t < total) xs[t] = x[t] * invOut[t / 69];
}

// ---------------- SpMM: wave-per-edge scatter-add (L2-resident) ----------------

template <int D>
__global__ __launch_bounds__(256) void spmm_scatter(const int* __restrict__ src,
                                                    const int* __restrict__ dst,
                                                    const float* __restrict__ in,
                                                    float* out, int ne) {
    int wid  = (int)(((long)blockIdx.x * blockDim.x + threadIdx.x) >> 5);
    int lane = threadIdx.x & 31;
    if (wid >= ne) return;
    const float* pi = in  + (long)src[wid] * D;
    float*       po = out + (long)dst[wid] * D;
    #pragma unroll
    for (int f = lane; f < D; f += 32)
        atomicAdd(&po[f], pi[f]);
}

// ---------------- WMMA GEMM: out[M,N] = relu?((A*rowScale) @ W) ----------------
// A: [M,K] f32 row-major.  W: [K,N] f32 row-major.  K <= KPAD <= 128, KPAD % 32 == 0.
// Block: 256 threads (8 waves), 128 rows per block; wave w owns rows 16w..16w+15.

template <int NT, int K, int KPAD, bool RELU>
__global__ __launch_bounds__(256) void gemm_wmma(const float* __restrict__ A,
                                                 const float* __restrict__ W,
                                                 const float* __restrict__ rowScale,
                                                 float* __restrict__ out, int M) {
    constexpr int MT = 128;
    __shared__ __align__(16) _Float16 As[MT * KPAD];      // As[m*KPAD + k]
    __shared__ __align__(16) _Float16 Bs[NT * KPAD];      // transposed: Bs[n*KPAD + k]

    const int tid  = threadIdx.x;
    const int row0 = blockIdx.x * MT;

    // Stage A tile (fold row scaling, convert f32 -> f16, zero-pad K..KPAD)
    for (int idx = tid; idx < MT * KPAD; idx += 256) {
        int m = idx / KPAD, k = idx - m * KPAD;
        int r = row0 + m;
        float v = 0.0f;
        if (r < M && k < K) v = A[(long)r * K + k] * rowScale[r];
        As[idx] = (_Float16)v;
    }
    // Stage W transposed (k-contiguous per output column), zero-padded
    for (int idx = tid; idx < NT * KPAD; idx += 256) {
        int n = idx / KPAD, k = idx - n * KPAD;
        float v = (k < K) ? W[(long)k * NT + n] : 0.0f;
        Bs[idx] = (_Float16)v;
    }
    __syncthreads();

    const int wid    = tid >> 5;
    const int lane   = tid & 31;
    const int laneHi = lane >> 4;    // 0: lanes 0-15, 1: lanes 16-31
    const int lmod   = lane & 15;
    const int m0     = wid * 16;

    constexpr int NTILES = NT / 16;
    v8f acc[NTILES];
    #pragma unroll
    for (int i = 0; i < NTILES; ++i)
        acc[i] = (v8f){0.f, 0.f, 0.f, 0.f, 0.f, 0.f, 0.f, 0.f};

    #pragma unroll
    for (int kb = 0; kb < KPAD; kb += 32) {
        // A fragment, contiguous k order (same order as B):
        // lane<16 holds K = kb+{0..15}; lane>=16 holds K = kb+{16..31}
        const int aoff = laneHi ? 16 : 0;
        const _Float16* pArow = &As[(m0 + lmod) * KPAD + kb + aoff];
        v8h alo = *reinterpret_cast<const v8h*>(pArow);
        v8h ahi = *reinterpret_cast<const v8h*>(pArow + 8);
        v16h afrag = __builtin_shufflevector(alo, ahi,
            0, 1, 2, 3, 4, 5, 6, 7, 8, 9, 10, 11, 12, 13, 14, 15);

        #pragma unroll
        for (int nt = 0; nt < NTILES; ++nt) {
            // B fragment, contiguous k order (same order as A):
            // lane = n (+16 for K upper half); element i -> K = kb + koff + i
            const int koff = laneHi ? 16 : 0;
            const _Float16* pBrow = &Bs[(nt * 16 + lmod) * KPAD + kb + koff];
            v8h blo = *reinterpret_cast<const v8h*>(pBrow);
            v8h bhi = *reinterpret_cast<const v8h*>(pBrow + 8);
            v16h bfrag = __builtin_shufflevector(blo, bhi,
                0, 1, 2, 3, 4, 5, 6, 7, 8, 9, 10, 11, 12, 13, 14, 15);

            acc[nt] = __builtin_amdgcn_wmma_f32_16x16x32_f16(
                /*neg_a=*/false, afrag, /*neg_b=*/false, bfrag,
                /*c_mod=*/(short)0, acc[nt], /*reuse_a=*/false, /*reuse_b=*/false);
        }
    }

    // C/D 16x16 f32 layout: VGPR r -> M = r + 8*laneHi, N = lane&15
    #pragma unroll
    for (int nt = 0; nt < NTILES; ++nt) {
        int col = nt * 16 + lmod;
        #pragma unroll
        for (int r = 0; r < 8; ++r) {
            int row = row0 + m0 + laneHi * 8 + r;
            if (row < M) {
                float v = acc[nt][r];
                if (RELU) v = fmaxf(v, 0.0f);
                out[(long)row * NT + col] = v;
            }
        }
    }
}

// ---------------- readout: relu(agg * invIn) then per-graph max ----------------
// Values are >= 0, so uint bit-pattern atomicMax == float max (d_out zeroed first).

__global__ __launch_bounds__(256) void readout_max(const float* __restrict__ agg,
                                                   const float* __restrict__ invIn,
                                                   const int* __restrict__ gid,
                                                   unsigned int* out, long total) {
    long t = (long)blockIdx.x * blockDim.x + threadIdx.x;
    if (t >= total) return;
    int n = (int)(t >> 6), f = (int)(t & 63);
    float v = fmaxf(agg[t] * invIn[n], 0.0f);
    atomicMax(&out[gid[n] * 64 + f], __float_as_uint(v));
}

// ---------------- launcher ----------------

extern "C" void kernel_launch(void* const* d_in, const int* in_sizes, int n_in,
                              void* d_out, int out_size, void* d_ws, size_t ws_size,
                              hipStream_t stream) {
    const float* x   = (const float*)d_in[0];   // [NN, 69]
    const float* W1  = (const float*)d_in[1];   // [69, 128]
    const float* W2  = (const float*)d_in[2];   // [128, 64]
    const int*   src = (const int*)d_in[3];     // [NE]
    const int*   dst = (const int*)d_in[4];     // [NE]
    const int*   gid = (const int*)d_in[5];     // [NN]

    const int NE = in_sizes[3];
    const int NN = in_sizes[5];

    // workspace layout (floats)
    float* invOut = (float*)d_ws;               // [NN]
    float* invIn  = invOut + NN;                // [NN]
    float* bufA   = invIn + NN;                 // [NN*69], reused as h2p [NN*64]
    float* bufB   = bufA + (long)NN * 69;       // [NN*69], reused as agg2 [NN*64]
    float* h1     = bufB + (long)NN * 69;       // [NN*128]

    const int B = 256;
    auto blocks = [](long n, int b) { return (unsigned)((n + b - 1) / b); };

    // degrees -> inverse sqrt norms (invOut/invIn are contiguous: one pass)
    zero_f32<<<blocks(2L * NN, B), B, 0, stream>>>(invOut, 2L * NN);
    degree_kernel<<<blocks(NE, B), B, 0, stream>>>(src, dst, invOut, invIn, NE);
    rsqrt_clip<<<blocks(2L * NN, B), B, 0, stream>>>(invOut, 2L * NN);

    // layer 1: xs = x * D_out^-1/2 ; agg1 = scatter-add(xs) ; h1 = relu((agg1*D_in^-1/2) @ W1)
    scale_rows69<<<blocks((long)NN * 69, B), B, 0, stream>>>(x, invOut, bufA, (long)NN * 69);
    zero_f32<<<blocks((long)NN * 69, B), B, 0, stream>>>(bufB, (long)NN * 69);
    spmm_scatter<69><<<blocks((long)NE * 32, B), B, 0, stream>>>(src, dst, bufA, bufB, NE);
    gemm_wmma<128, 69, 96, true><<<blocks(NN, 128), B, 0, stream>>>(bufB, W1, invIn, h1, NN);

    // layer 2 (W applied before SpMM; linear ops commute):
    // h2p = (h1 * D_out^-1/2) @ W2 ; agg2 = scatter-add(h2p) ; node = relu(agg2 * D_in^-1/2)
    gemm_wmma<64, 128, 128, false><<<blocks(NN, 128), B, 0, stream>>>(h1, W2, invOut, bufA, NN);
    zero_f32<<<blocks((long)NN * 64, B), B, 0, stream>>>(bufB, (long)NN * 64);
    spmm_scatter<64><<<blocks((long)NE * 32, B), B, 0, stream>>>(src, dst, bufA, bufB, NE);

    // readout: per-graph max (values >= 0, uint-ordered atomicMax)
    zero_f32<<<blocks(NGRAPHS * 64, B), B, 0, stream>>>((float*)d_out, NGRAPHS * 64);
    readout_max<<<blocks((long)NN * 64, B), B, 0, stream>>>(bufB, invIn, gid,
                                                            (unsigned int*)d_out, (long)NN * 64);
}
